// Attention_82325933130454
// MI455X (gfx1250) — compile-verified, matrix operand order checked by reading, not run on hardware
//
#include <hip/hip_runtime.h>

typedef __bf16 bf16;
typedef __attribute__((ext_vector_type(16))) __bf16 v16bf;
typedef __attribute__((ext_vector_type(8)))  __bf16 v8bf;
typedef __attribute__((ext_vector_type(8)))  float   v8f;
typedef __attribute__((ext_vector_type(4)))  float   v4f;
typedef __attribute__((ext_vector_type(4)))  unsigned int u32x4;
typedef __attribute__((ext_vector_type(8)))  unsigned int u32x8;

#define DIM   4096
#define NH    32
#define HD    128
#define BSZ   2
#define SEQ   2048
#define ROWS  (BSZ*SEQ)
#define NK    (DIM/32)

__device__ __forceinline__ v8f wmma_bf16f32(v16bf a, v16bf b, v8f c) {
  // (neg_a, A, neg_b, B, c_mod, C, reuse_a, reuse_b)
  return __builtin_amdgcn_wmma_f32_16x16x32_bf16(false, a, false, b, (short)0, c, false, false);
}

// Load one 16x32 (MxK) A-style fragment (B uses the mirrored layout with N in
// place of M) from a K-contiguous, row-strided bf16 source.
__device__ __forceinline__ v16bf ldfrag_bf16(const bf16* p, int rs, int lane) {
  const int r  = lane & 15;
  const int ko = (lane >> 4) << 3;
  const bf16* q = p + (size_t)r * rs + ko;
  v8bf lo = *(const v8bf*)(q);        // K pairs 0..7   (this half)
  v8bf hi = *(const v8bf*)(q + 16);   // K pairs 16..23 (this half)
  v16bf f;
#pragma unroll
  for (int i = 0; i < 8; ++i) { f[i] = lo[i]; f[i + 8] = hi[i]; }
  return f;
}

// Same fragment, fp32 source, convert to bf16 in registers.
__device__ __forceinline__ v16bf ldfrag_f32(const float* p, int rs, int lane) {
  const int r  = lane & 15;
  const int ko = (lane >> 4) << 3;
  const float* q = p + (size_t)r * rs + ko;
  v4f a = *(const v4f*)(q);
  v4f b = *(const v4f*)(q + 4);
  v4f c = *(const v4f*)(q + 16);
  v4f d = *(const v4f*)(q + 20);
  v16bf f;
#pragma unroll
  for (int i = 0; i < 4; ++i) {
    f[i]      = (bf16)a[i];
    f[i + 4]  = (bf16)b[i];
    f[i + 8]  = (bf16)c[i];
    f[i + 12] = (bf16)d[i];
  }
  return f;
}

// ---------------------------------------------------------------------------
// MODE 0: Q  = x @ wq^T, RoPE, bf16 -> ws
// MODE 1: K  = x @ wk^T, RoPE, fp32 -> cache_k, bf16 -> ws
// MODE 2: V  = x @ wv^T,        fp32 -> cache_v, bf16 -> ws
// MODE 3: out= attn @ wo^T (A is bf16 from ws), fp32 -> out
// Double-buffered LDS weight tile: global loads for step kt+1 are issued
// before the WMMA chain on step kt.
// ---------------------------------------------------------------------------
#define BK 40  // padded LDS K-stride (elements), 80B = 16B aligned, conflict-free
template <int MODE>
__global__ __launch_bounds__(256) void gemm_rope(
    const float* __restrict__ X, const bf16* __restrict__ Xb,
    const float* __restrict__ W,
    float* __restrict__ OutF, bf16* __restrict__ OutB,
    const float* __restrict__ cosT, const float* __restrict__ sinT) {
  __shared__ bf16 Bs[2][128 * BK];

  const int tid   = threadIdx.x;
  const int lane  = tid & 31;
  const int wave  = tid >> 5;
  const int mBase = blockIdx.x * 128 + wave * 16;
  const int nBase = blockIdx.y * 128;

  v8f acc[8];
#pragma unroll
  for (int j = 0; j < 8; ++j)
#pragma unroll
    for (int r = 0; r < 8; ++r) acc[j][r] = 0.f;

  const int stgRow = tid >> 1;            // weight-tile staging coords
  const int stgK   = (tid & 1) * 16;
  const float* wrow = W + (size_t)(nBase + stgRow) * DIM + stgK;

  // prologue: stage tile 0 into buffer 0, load A fragment 0
  {
    v4f f0 = *(const v4f*)(wrow);
    v4f f1 = *(const v4f*)(wrow + 4);
    v4f f2 = *(const v4f*)(wrow + 8);
    v4f f3 = *(const v4f*)(wrow + 12);
    bf16* dst = &Bs[0][stgRow * BK + stgK];
#pragma unroll
    for (int i = 0; i < 4; ++i) {
      dst[i]      = (bf16)f0[i];
      dst[i + 4]  = (bf16)f1[i];
      dst[i + 8]  = (bf16)f2[i];
      dst[i + 12] = (bf16)f3[i];
    }
  }
  v16bf afrag;
  if (MODE == 3) afrag = ldfrag_bf16(Xb + (size_t)mBase * DIM, DIM, lane);
  else           afrag = ldfrag_f32(X + (size_t)mBase * DIM, DIM, lane);
  __syncthreads();

  for (int kt = 0; kt < NK; ++kt) {
    const int  cur     = kt & 1;
    const bool hasNext = (kt + 1) < NK;

    // issue next tile's global loads early (latency hidden by WMMA chain)
    v4f f0, f1, f2, f3;
    v16bf afragN;
    if (hasNext) {
      const float* src = wrow + (kt + 1) * 32;
      f0 = *(const v4f*)(src);
      f1 = *(const v4f*)(src + 4);
      f2 = *(const v4f*)(src + 8);
      f3 = *(const v4f*)(src + 12);
      if (MODE == 3)
        afragN = ldfrag_bf16(Xb + (size_t)mBase * DIM + (kt + 1) * 32, DIM, lane);
      else
        afragN = ldfrag_f32(X + (size_t)mBase * DIM + (kt + 1) * 32, DIM, lane);
      if (kt + 2 < NK) __builtin_prefetch(wrow + (kt + 2) * 32, 0, 0);
    }

    // compute on current buffer: preload all 8 B fragments, then WMMA chain
    v16bf bfrag[8];
#pragma unroll
    for (int j = 0; j < 8; ++j)
      bfrag[j] = ldfrag_bf16(&Bs[cur][(j * 16) * BK], BK, lane);
#pragma unroll
    for (int j = 0; j < 8; ++j)
      acc[j] = wmma_bf16f32(afrag, bfrag[j], acc[j]);

    // finish staging the next buffer
    if (hasNext) {
      bf16* dst = &Bs[1 - cur][stgRow * BK + stgK];
#pragma unroll
      for (int i = 0; i < 4; ++i) {
        dst[i]      = (bf16)f0[i];
        dst[i + 4]  = (bf16)f1[i];
        dst[i + 8]  = (bf16)f2[i];
        dst[i + 12] = (bf16)f3[i];
      }
      afrag = afragN;
    }
    __syncthreads();
  }

  // Epilogue: C element (m,n) -> lane = n + 16*(m>=8), vgpr = m&7.
  const int half = lane >> 4;
  const int nIdx = lane & 15;
#pragma unroll
  for (int j = 0; j < 8; ++j) {
    const int nG = nBase + j * 16 + nIdx;
#pragma unroll
    for (int r = 0; r < 8; ++r) {
      const int m = mBase + r + 8 * half;
      float v = acc[j][r];
      if (MODE == 0 || MODE == 1) {  // fused RoPE (partner is the xor-1 lane)
        const int   d    = nG & (HD - 1);
        const int   srow = m & (SEQ - 1);
        const float c    = cosT[srow * (HD / 2) + (d >> 1)];
        const float s    = sinT[srow * (HD / 2) + (d >> 1)];
        const float o    = __shfl_xor(v, 1, 32);
        v = (d & 1) ? (o * s + v * c) : (v * c - o * s);
      }
      const size_t idx = (size_t)m * DIM + nG;
      if (MODE == 1 || MODE == 2 || MODE == 3) OutF[idx] = v;
      if (MODE != 3) OutB[idx] = (bf16)v;
    }
  }
}

// ---------------------------------------------------------------------------
// Flash attention: one block per (b, h, 128-row q block), 8 waves.
// K tiles double-buffered and fetched by the Tensor Data Mover (descriptor-
// side LDS padding -> stride 136); the TDM for block kb+1 runs concurrently
// with compute on block kb. V prefetched to registers, scattered (transposed)
// between barriers.
// ---------------------------------------------------------------------------
#define KS 136  // padded LDS stride (elements); 272B, 16B aligned
__global__ __launch_bounds__(256) void attn(
    const bf16* __restrict__ Qb, const bf16* __restrict__ Kb,
    const bf16* __restrict__ Vb, bf16* __restrict__ Ob) {
  __shared__ bf16 Ksh[2][128 * KS];     // [kpos][d], filled by TDM
  __shared__ bf16 Vt[HD * KS];          // [d][kpos] (transposed)
  __shared__ bf16 Ps[8][16 * KS];       // per-wave P tile [m][kpos]

  const int b    = blockIdx.z;
  const int h    = blockIdx.y;
  const int qblk = blockIdx.x;
  const int tid  = threadIdx.x;
  const int lane = tid & 31;
  const int wave = tid >> 5;
  const float scale = 0.08838834764831845f;  // 1/sqrt(128)

  const size_t headOff = (size_t)h * HD;
  const bf16* Qbase = Qb + ((size_t)(b * SEQ + qblk * 128 + wave * 16)) * DIM + headOff;

  v16bf qf[4];
#pragma unroll
  for (int c = 0; c < 4; ++c) qf[c] = ldfrag_bf16(Qbase + c * 32, DIM, lane);

  v8f o[8];
  float mrun[8], lrun[8];
#pragma unroll
  for (int j = 0; j < 8; ++j)
#pragma unroll
    for (int r = 0; r < 8; ++r) o[j][r] = 0.f;
#pragma unroll
  for (int r = 0; r < 8; ++r) { mrun[r] = -3.0e38f; lrun[r] = 0.f; }

  const int half  = lane >> 4;
  const int nIdx  = lane & 15;
  const int mRow0 = qblk * 128 + wave * 16 + 8 * half;  // row of vgpr r = mRow0+r

  // --- constant TDM descriptor pieces ---
  // group1: wg_mask=0 | data_size=2B | pad_enable, pad_interval=64 DWORDs,
  //         pad_amount=4 DWORDs (-> row stride 128+8 elements = KS)
  u32x8 g1;
  g1[0] = (1u << 16) | (1u << 20) | (5u << 22) | (3u << 25);
  g1[1] = (unsigned)DIM << 16;        // tensor_dim0[15:0] into bits[63:48]
  g1[2] = (unsigned)(ROWS) << 16;     // tensor_dim0 hi=0; tensor_dim1 lo
  g1[3] = 128u << 16;                 // tensor_dim1 hi=0; tile_dim0 = 128
  g1[4] = 128u;                       // tile_dim1 = 128; tile_dim2 = 0
  g1[5] = (unsigned)DIM;              // tensor_dim0_stride = 4096 elements
  g1[6] = 0u;
  g1[7] = 0u;
  u32x4 g2; g2[0]=0u; g2[1]=0u; g2[2]=0u; g2[3]=0u;
  u32x4 g3; g3[0]=0u; g3[1]=0u; g3[2]=0u; g3[3]=0u;
  const unsigned ldsK0 = (unsigned)(uintptr_t)(void*)&Ksh[0][0];
  const unsigned ldsK1 = (unsigned)(uintptr_t)(void*)&Ksh[1][0];

  const int vRow = tid >> 1;
  const int vD0  = (tid & 1) * 64;

  auto issueTDM = [&](int kb, int buf) {
    if (wave == 0) {
      const unsigned long long ga =
          (unsigned long long)(uintptr_t)(Kb + ((size_t)(b * SEQ + kb * 128)) * DIM + headOff);
      u32x4 g0;
      g0[0] = 1u;                                        // count=1, no gather
      g0[1] = buf ? ldsK1 : ldsK0;                       // lds_addr
      g0[2] = (unsigned)(ga & 0xffffffffu);              // global_addr[31:0]
      g0[3] = (unsigned)((ga >> 32) & 0x1ffffffu) | (2u << 30);  // addr[56:32] | type=2
      asm volatile("tensor_load_to_lds %0, %1, %2, %3"
                   :: "s"(g0), "s"(g1), "s"(g2), "s"(g3)
                   : "memory");
    }
  };
  auto loadV = [&](int kb, v8bf vr[8]) {
    const bf16* src = Vb + ((size_t)(b * SEQ + kb * 128 + vRow)) * DIM + headOff + vD0;
#pragma unroll
    for (int i = 0; i < 8; ++i) vr[i] = *(const v8bf*)(src + i * 8);
  };
  auto scatterV = [&](const v8bf vr[8]) {
#pragma unroll
    for (int i = 0; i < 8; ++i)
#pragma unroll
      for (int jj = 0; jj < 8; ++jj) Vt[(vD0 + i * 8 + jj) * KS + vRow] = vr[i][jj];
  };

  // prologue: K(0) via TDM, V(0) via registers
  v8bf vr[8];
  issueTDM(0, 0);
  loadV(0, vr);
  scatterV(vr);
  if (wave == 0) __builtin_amdgcn_s_wait_tensorcnt(0);
  __syncthreads();

  for (int kb = 0; kb <= qblk; ++kb) {
    const int  cur     = kb & 1;
    const bool hasNext = (kb + 1) <= qblk;

    if (hasNext) {           // overlap next K DMA + V loads with compute
      issueTDM(kb + 1, 1 - cur);
      loadV(kb + 1, vr);
    }

    const bool diag = (kb == qblk);

    // S = Q K^T * scale (+ causal mask on diagonal block)
    v8f st[8];
#pragma unroll
    for (int j = 0; j < 8; ++j) {
      v16bf kf[4];
#pragma unroll
      for (int c = 0; c < 4; ++c)
        kf[c] = ldfrag_bf16(&Ksh[cur][(j * 16) * KS] + c * 32, KS, lane);
      v8f s;
#pragma unroll
      for (int r = 0; r < 8; ++r) s[r] = 0.f;
#pragma unroll
      for (int c = 0; c < 4; ++c) s = wmma_bf16f32(qf[c], kf[c], s);
#pragma unroll
      for (int r = 0; r < 8; ++r) {
        float v = s[r] * scale;
        if (diag) {
          const int nG = kb * 128 + j * 16 + nIdx;
          if (nG > mRow0 + r) v += -1.0e9f;
        }
        s[r] = v;
      }
      st[j] = s;
    }

    // online softmax: new row max (reduce across j tiles + 16-lane half)
    float mnew[8];
#pragma unroll
    for (int r = 0; r < 8; ++r) {
      float mx = mrun[r];
#pragma unroll
      for (int j = 0; j < 8; ++j) mx = fmaxf(mx, st[j][r]);
#pragma unroll
      for (int off = 8; off; off >>= 1) mx = fmaxf(mx, __shfl_xor(mx, off, 16));
      mnew[r] = mx;
    }

    // P = exp(S - mnew) -> per-wave LDS tile; accumulate row sums
    float ladd[8];
#pragma unroll
    for (int r = 0; r < 8; ++r) ladd[r] = 0.f;
#pragma unroll
    for (int j = 0; j < 8; ++j) {
#pragma unroll
      for (int r = 0; r < 8; ++r) {
        const float p = __expf(st[j][r] - mnew[r]);
        ladd[r] += p;
        Ps[wave][(r + 8 * half) * KS + j * 16 + nIdx] = (bf16)p;
      }
    }
#pragma unroll
    for (int r = 0; r < 8; ++r) {
      float l = ladd[r];
#pragma unroll
      for (int off = 8; off; off >>= 1) l += __shfl_xor(l, off, 16);
      const float alpha = __expf(mrun[r] - mnew[r]);
      lrun[r] = lrun[r] * alpha + l;
      mrun[r] = mnew[r];
#pragma unroll
      for (int j = 0; j < 8; ++j) o[j][r] *= alpha;
    }

    asm volatile("s_wait_dscnt 0" ::: "memory");  // P writes visible to this wave

    // O += P @ V
    v16bf pf[4];
#pragma unroll
    for (int c = 0; c < 4; ++c)
      pf[c] = ldfrag_bf16(&Ps[wave][0] + c * 32, KS, lane);
#pragma unroll
    for (int j = 0; j < 8; ++j) {
      v16bf vf[4];
#pragma unroll
      for (int c = 0; c < 4; ++c)
        vf[c] = ldfrag_bf16(Vt + (j * 16) * KS + c * 32, KS, lane);
#pragma unroll
      for (int c = 0; c < 4; ++c) o[j] = wmma_bf16f32(pf[c], vf[c], o[j]);
    }

    __syncthreads();          // all waves done reading Vt / Ksh[cur]
    if (hasNext) {
      scatterV(vr);           // single Vt buffer: refill between barriers
      if (wave == 0) __builtin_amdgcn_s_wait_tensorcnt(0);
    }
    __syncthreads();          // new Vt + TDM K tile visible
  }

  // normalize and store bf16 attention output to ws
  const int qG = qblk * 128 + wave * 16;
#pragma unroll
  for (int j = 0; j < 8; ++j) {
#pragma unroll
    for (int r = 0; r < 8; ++r) {
      const int m = r + 8 * half;
      const float v = o[j][r] / lrun[r];
      const size_t idx = ((size_t)(b * SEQ + qG + m)) * DIM + headOff + j * 16 + nIdx;
      Ob[idx] = (bf16)v;
    }
  }
}

extern "C" void kernel_launch(void* const* d_in, const int* in_sizes, int n_in,
                              void* d_out, int out_size, void* d_ws, size_t ws_size,
                              hipStream_t stream) {
  (void)in_sizes; (void)n_in; (void)out_size; (void)ws_size;
  const float* x    = (const float*)d_in[0];
  const float* fcos = (const float*)d_in[1];
  const float* fsin = (const float*)d_in[2];
  const float* wq   = (const float*)d_in[7];
  const float* wk   = (const float*)d_in[8];
  const float* wv   = (const float*)d_in[9];
  const float* wo   = (const float*)d_in[10];

  float* out     = (float*)d_out;                 // (B,S,DIM)
  float* cache_k = out + (size_t)ROWS * DIM;      // (B,S,NH,HD) == (B,S,DIM) flat
  float* cache_v = cache_k + (size_t)ROWS * DIM;

  bf16* qb = (bf16*)d_ws;
  bf16* kb = qb + (size_t)ROWS * DIM;
  bf16* vb = kb + (size_t)ROWS * DIM;
  bf16* ab = vb + (size_t)ROWS * DIM;

  dim3 g(ROWS / 128, DIM / 128), blk(256);
  gemm_rope<0><<<g, blk, 0, stream>>>(x, nullptr, wq, nullptr, qb, fcos, fsin);
  gemm_rope<1><<<g, blk, 0, stream>>>(x, nullptr, wk, cache_k, kb, fcos, fsin);
  gemm_rope<2><<<g, blk, 0, stream>>>(x, nullptr, wv, cache_v, vb, fcos, fsin);
  attn<<<dim3(SEQ / 128, NH, BSZ), blk, 0, stream>>>(qb, kb, vb, ab);
  gemm_rope<3><<<g, blk, 0, stream>>>(nullptr, ab, wo, out, nullptr, fcos, fsin);
}
